// ASE_block_67954972557764
// MI455X (gfx1250) — matrix operand order repaired and checked by reading, not compiled, not run on hardware
//
#include <hip/hip_runtime.h>

// ---------------------------------------------------------------------------
// SeaFormer-style squeeze-axial attention block on MI455X (gfx1250, wave32).
// All GEMMs use v_wmma_f32_16x16x32_f16 with f32 accumulation.
// Weight tiles are staged into LDS with GLOBAL_LOAD_ASYNC_TO_LDS_B128.
// ---------------------------------------------------------------------------

#define BATCH  16
#define DIM    256
#define HH     64
#define WW     64
#define HWSZ   (HH * WW)          // 4096
#define NH_KD  128
#define DH     256
#define CQKV   512                // nh_kd + nh_kd + dh
#define HEADS  8
#define KD     16
#define DHEAD  32                 // DH / HEADS
#define POS    16

typedef __attribute__((ext_vector_type(16))) _Float16 v16h;
typedef __attribute__((ext_vector_type(8)))  _Float16 v8h;
typedef __attribute__((ext_vector_type(8)))  float    v8f;

// ---- WMMA fragment helpers (CDNA5 wave32 layouts, 16x16x32 f16) ----------
// A 16x32 f16: lanes 0-15 row M=lane, K runs [8*hk, +8) and [8*hk+16, +8)
//              lanes16-31 row M=lane-16, hk = 1 adds K offset 8.
// B 32x16 f16: lane N = lane&15, K runs [8*hk, +8) and [8*hk+16, +8) (same as A).
__device__ __forceinline__ v16h mk16(const _Float16* p0, const _Float16* p1) {
  v8h lo = *(const v8h*)p0;
  v8h hi = *(const v8h*)p1;
  v16h r;
#pragma unroll
  for (int i = 0; i < 8; ++i) { r[i] = lo[i]; r[i + 8] = hi[i]; }
  return r;
}
__device__ __forceinline__ v16h fragA16(const _Float16* row, int hk, int kb) {
  const _Float16* p = row + kb + 8 * hk;
  return mk16(p, p + 16);
}
__device__ __forceinline__ v16h fragB16(const _Float16* row, int hk, int kb) {
  const _Float16* p = row + kb + 8 * hk;
  return mk16(p, p + 16);
}
__device__ __forceinline__ v8f wmma32(v16h a, v16h b, v8f c) {
  return __builtin_amdgcn_wmma_f32_16x16x32_f16(false, a, false, b,
                                                (short)0, c, false, false);
}

// ---- CDNA5 async global->LDS copy (ASYNCcnt-tracked, bypasses VGPRs) -----
// LDS dest address = low 32 bits of the generic shared pointer (ISA: LDS
// aperture addresses map to LDS by truncation to addr[31:0]).
__device__ __forceinline__ void async_copy16(const void* g, void* l) {
  unsigned lds = (unsigned)(unsigned long long)l;
  unsigned long long ga = (unsigned long long)g;
  asm volatile("global_load_async_to_lds_b128 %0, %1, off"
               :: "v"(lds), "v"(ga) : "memory");
}
__device__ __forceinline__ void async_wait0() {
  asm volatile("s_wait_asynccnt 0x0" ::: "memory");
}

// ---------------------------------------------------------------------------
// Prep kernels
// ---------------------------------------------------------------------------
__global__ void k_foldW(_Float16* __restrict__ dst, const float* __restrict__ W,
                        const float* __restrict__ s, int N, int K) {
  int i = blockIdx.x * 256 + threadIdx.x;
  if (i >= N * K) return;
  dst[i] = (_Float16)(W[i] * s[i / K]);
}

__global__ void k_copy(float* __restrict__ dst, const float* __restrict__ src, int n) {
  int i = blockIdx.x * 256 + threadIdx.x;
  if (i < n) dst[i] = src[i];
}

__global__ void k_mulvec(float* __restrict__ dst, const float* __restrict__ a,
                         const float* __restrict__ b, int n) {
  int i = blockIdx.x * 256 + threadIdx.x;
  if (i < n) dst[i] = a[i] * b[i];
}

// linear interp (C=128, POS=16) -> (128, 64), align_corners=False
__global__ void k_pe(float* __restrict__ dst, const float* __restrict__ src) {
  int i = blockIdx.x * 256 + threadIdx.x;
  if (i >= NH_KD * 64) return;
  int c = i >> 6, x = i & 63;
  float pos = (x + 0.5f) * 0.25f - 0.5f;
  pos = fminf(fmaxf(pos, 0.0f), 15.0f);
  int i0 = (int)floorf(pos);
  int i1 = i0 + 1 > 15 ? 15 : i0 + 1;
  float w = pos - (float)i0;
  dst[i] = src[c * POS + i0] * (1.0f - w) + src[c * POS + i1] * w;
}

// ---------------------------------------------------------------------------
// Row / column means of qkv: one block per (b, c) 64x64 tile.
// ---------------------------------------------------------------------------
__global__ __launch_bounds__(256) void k_means(const float* __restrict__ qkv,
                                               float* __restrict__ rowm,
                                               float* __restrict__ colm) {
  int bc = blockIdx.x;                       // 0 .. B*CQKV-1
  const float* tile = qkv + (long long)bc * HWSZ;
  __shared__ float sm[HH][WW + 1];
  int tid = threadIdx.x;
#pragma unroll
  for (int u = 0; u < 16; ++u) {
    int e = tid + u * 256;
    sm[e >> 6][e & 63] = tile[e];
  }
  __syncthreads();
  if (tid < 64) {                            // row mean (over w), h = tid
    float s = 0.f;
#pragma unroll
    for (int w = 0; w < WW; ++w) s += sm[tid][w];
    rowm[(long long)bc * 64 + tid] = s * (1.0f / WW);
  } else if (tid < 128) {                    // col mean (over h), w = tid-64
    int w = tid - 64;
    float s = 0.f;
#pragma unroll
    for (int h = 0; h < HH; ++h) s += sm[h][w];
    colm[(long long)bc * 64 + w] = s * (1.0f / HH);
  }
}

// ---------------------------------------------------------------------------
// Axial attention: one block (4 waves, 128 threads) per (batch, head).
// S = (q+pe)(64x16) @ (k+pe)^T(16x64) * 0.25 ; softmax rows ; O = P @ V(64x32)
// means layout: [B][512][64] (q:0-127, k:128-255, v:256-511)
// out: xpre [B][256][64] (pre-ReLU)
// ---------------------------------------------------------------------------
__global__ __launch_bounds__(128) void k_axial(const float* __restrict__ means,
                                               const float* __restrict__ peq,
                                               const float* __restrict__ pek,
                                               float* __restrict__ xpre) {
  int b = blockIdx.x >> 3;
  int head = blockIdx.x & 7;
  __shared__ __align__(16) _Float16 aq[64][40];   // q rows, K padded 16->32
  __shared__ __align__(16) _Float16 bk[64][40];   // k^T rows (N-major), K padded
  __shared__ __align__(16) _Float16 vv[32][72];   // V^T: [d][j]
  __shared__ __align__(16) _Float16 pp[64][72];   // softmax probs f16

  int tid = threadIdx.x;
  const long long mb = (long long)b * CQKV * 64;
#pragma unroll
  for (int u = 0; u < 16; ++u) {                  // 2048 entries each
    int e = tid + u * 128;
    int i = e >> 5, kd = e & 31;
    float vq = 0.f, vk = 0.f;
    if (kd < KD) {
      int c = head * KD + kd;
      vq = means[mb + (long long)c * 64 + i] + peq[c * 64 + i];
      vk = means[mb + (long long)(NH_KD + c) * 64 + i] + pek[c * 64 + i];
    }
    aq[i][kd] = (_Float16)vq;
    bk[i][kd] = (_Float16)vk;
  }
#pragma unroll
  for (int u = 0; u < 16; ++u) {                  // V: 32x64
    int e = tid + u * 128;
    int d = e >> 6, j = e & 63;
    int c = 2 * NH_KD + head * DHEAD + d;
    vv[d][j] = (_Float16)means[mb + (long long)c * 64 + j];
  }
  __syncthreads();

  int wv = tid >> 5, lane = tid & 31;
  int m0 = wv * 16;
  int hk = lane >> 4, l15 = lane & 15;

  // ---- S = q @ k^T (K=32, padded) ----
  v8f s[4] = {};
  {
    v16h af = fragA16(&aq[m0 + l15][0], hk, 0);
#pragma unroll
    for (int t = 0; t < 4; ++t) {
      v16h bf = fragB16(&bk[t * 16 + l15][0], hk, 0);
      s[t] = wmma32(af, bf, s[t]);
    }
  }
  // ---- softmax over rows (each row lives in one 16-lane half) ----
#pragma unroll
  for (int t = 0; t < 4; ++t)
#pragma unroll
    for (int r = 0; r < 8; ++r) s[t][r] *= 0.25f;       // KEY_DIM^-0.5

#pragma unroll
  for (int r = 0; r < 8; ++r) {
    float mx = -1e30f;
#pragma unroll
    for (int t = 0; t < 4; ++t) mx = fmaxf(mx, s[t][r]);
#pragma unroll
    for (int off = 1; off < 16; off <<= 1) mx = fmaxf(mx, __shfl_xor(mx, off, 32));
    float sum = 0.f;
#pragma unroll
    for (int t = 0; t < 4; ++t) {
      float e = __expf(s[t][r] - mx);
      s[t][r] = e;
      sum += e;
    }
#pragma unroll
    for (int off = 1; off < 16; off <<= 1) sum += __shfl_xor(sum, off, 32);
    float inv = 1.0f / sum;
    int row = m0 + r + 8 * hk;
#pragma unroll
    for (int t = 0; t < 4; ++t) pp[row][t * 16 + l15] = (_Float16)(s[t][r] * inv);
  }
  __syncthreads();

  // ---- O = P @ V  (K=64 -> 2 steps, N=32 -> 2 tiles) ----
  v8f o[2] = {};
#pragma unroll
  for (int ks = 0; ks < 2; ++ks) {
    v16h af = fragA16(&pp[m0 + l15][0], hk, ks * 32);
#pragma unroll
    for (int t = 0; t < 2; ++t) {
      v16h bf = fragB16(&vv[t * 16 + l15][0], hk, ks * 32);
      o[t] = wmma32(af, bf, o[t]);
    }
  }
#pragma unroll
  for (int t = 0; t < 2; ++t) {
    int c = head * DHEAD + t * 16 + l15;
    long long base = ((long long)b * DH + c) * 64 + m0 + 8 * hk;
#pragma unroll
    for (int r = 0; r < 8; ++r) xpre[base + r] = o[t][r];
  }
}

// ---------------------------------------------------------------------------
// Generic fused GEMM: out[b][n][s] = epi( sum_k W[n][k] * pre(in[b][k][s]) + bias[n] )
//   pre: optional per-channel affine (preScale/preBias), optional axial adds
//        (addR[b][k][s/Wd] + addC[b][k][s%Wd]), optional ReLU.
//   epi: plain, or h_sigmoid * gate[b][n][s] (final projection).
// Block: 256 threads = 8 waves (2 M-waves x 4 N-waves); tile M=64, N=128.
// B (weight) tiles are copied global->LDS with async-to-LDS b128 ops.
// ---------------------------------------------------------------------------
__global__ __launch_bounds__(256) void k_gemm(
    const float* __restrict__ in, long long inBatchStride, int K, int S,
    const float* __restrict__ preScale, const float* __restrict__ preBias,
    int inRelu, const float* __restrict__ addR, const float* __restrict__ addC,
    const _Float16* __restrict__ Wt, const float* __restrict__ bias, int Ntot,
    const float* __restrict__ gate, float* __restrict__ out) {
  __shared__ __align__(16) _Float16 aS[64][40];
  __shared__ __align__(16) _Float16 bS[128][40];

  int mtPerBatch = S / 64;
  int bIdx = blockIdx.x / mtPerBatch;
  int s0 = (blockIdx.x % mtPerBatch) * 64;
  int n0 = blockIdx.y * 128;
  int tid = threadIdx.x;
  int wv = tid >> 5, lane = tid & 31;
  int mw = wv & 1, nw = wv >> 1;
  int hk = lane >> 4, l15 = lane & 15;

  v8f acc[2][2] = {};
  const float* inB = in + (long long)bIdx * inBatchStride;

  for (int k0 = 0; k0 < K; k0 += 32) {
    {  // ---- stage B: async global->LDS, 512 x 16B chunks, no VGPR round-trip
#pragma unroll
      for (int c2 = 0; c2 < 2; ++c2) {
        int chunk = tid + 256 * c2;          // 0..511
        int n = chunk >> 2, part = chunk & 3; // 4 x 16B per 32-half row
        async_copy16(Wt + (long long)(n0 + n) * K + k0 + part * 8,
                     &bS[n][part * 8]);
      }
    }
    {  // ---- stage A (f32 -> f16), fused input transform ----
      int kk = tid >> 3;            // 0..31
      int part = tid & 7;           // 0..7 -> 8 consecutive pixels each
      int kG = k0 + kk;
      const float* src = inB + (long long)kG * S + s0 + part * 8;
      float ps = preScale ? preScale[kG] : 1.0f;
      float pb = preBias ? preBias[kG] : 0.0f;
#pragma unroll
      for (int u = 0; u < 8; ++u) {
        int p = part * 8 + u;
        float v = src[u];
        if (addR) {
          int sIdx = s0 + p;
          v += addR[((long long)bIdx * K + kG) * (S / WW) + (sIdx >> 6)]
             + addC[((long long)bIdx * K + kG) * WW + (sIdx & (WW - 1))];
        }
        v = v * ps + pb;
        if (inRelu) v = fmaxf(v, 0.0f);
        aS[p][kk] = (_Float16)v;
      }
    }
    async_wait0();                  // async B-copies landed in LDS
    __syncthreads();                // + everyone's ds_stores visible
#pragma unroll
    for (int a = 0; a < 2; ++a) {
      v16h af = fragA16(&aS[mw * 32 + a * 16 + l15][0], hk, 0);
#pragma unroll
      for (int bb = 0; bb < 2; ++bb) {
        v16h bf = fragB16(&bS[nw * 32 + bb * 16 + l15][0], hk, 0);
        acc[a][bb] = wmma32(af, bf, acc[a][bb]);
      }
    }
    __syncthreads();                // protect LDS before next iteration's copies
  }

#pragma unroll
  for (int a = 0; a < 2; ++a) {
#pragma unroll
    for (int bb = 0; bb < 2; ++bb) {
      int n = n0 + nw * 32 + bb * 16 + l15;
      float bv = bias[n];
      long long row = ((long long)bIdx * Ntot + n) * S;
      int sb = s0 + mw * 32 + a * 16 + 8 * hk;
#pragma unroll
      for (int r = 0; r < 8; ++r) {
        float v = acc[a][bb][r] + bv;
        if (gate) {
          v = fminf(fmaxf(v + 3.0f, 0.0f), 6.0f) * (1.0f / 6.0f);
          v *= gate[row + sb + r];
        }
        out[row + sb + r] = v;
      }
    }
  }
}

// ---------------------------------------------------------------------------
// Host-side orchestration
// ---------------------------------------------------------------------------
extern "C" void kernel_launch(void* const* d_in, const int* in_sizes, int n_in,
                              void* d_out, int out_size, void* d_ws, size_t ws_size,
                              hipStream_t stream) {
  (void)in_sizes; (void)n_in; (void)out_size; (void)ws_size;

  const float* x    = (const float*)d_in[0];
  const float* Wq   = (const float*)d_in[1];
  const float* sq   = (const float*)d_in[2];
  const float* bq   = (const float*)d_in[3];
  const float* Wk   = (const float*)d_in[4];
  const float* sk   = (const float*)d_in[5];
  const float* bk_  = (const float*)d_in[6];
  const float* Wv   = (const float*)d_in[7];
  const float* sv   = (const float*)d_in[8];
  const float* bv   = (const float*)d_in[9];
  const float* pe_rq = (const float*)d_in[10];
  const float* pe_rk = (const float*)d_in[11];
  const float* pe_cq = (const float*)d_in[12];
  const float* pe_ck = (const float*)d_in[13];
  const float* wdw  = (const float*)d_in[14];
  const float* sdw  = (const float*)d_in[15];
  const float* bdw  = (const float*)d_in[16];
  const float* Wpw  = (const float*)d_in[17];
  const float* spw  = (const float*)d_in[18];
  const float* bpw  = (const float*)d_in[19];
  const float* Wr   = (const float*)d_in[20];
  const float* sr   = (const float*)d_in[21];
  const float* br   = (const float*)d_in[22];
  const float* Wc   = (const float*)d_in[23];
  const float* sc   = (const float*)d_in[24];
  const float* bc   = (const float*)d_in[25];
  const float* Wp   = (const float*)d_in[26];
  const float* sp   = (const float*)d_in[27];
  const float* bp   = (const float*)d_in[28];

  // -------- workspace carve-up --------
  char* w = (char*)d_ws;
  size_t off = 0;
  auto take = [&](size_t nbytes) -> void* {
    void* p = w + off;
    off = (off + nbytes + 255) & ~(size_t)255;
    return p;
  };
  _Float16* WqkvH = (_Float16*)take((size_t)CQKV * DIM * 2);
  _Float16* WpwH  = (_Float16*)take((size_t)DIM * (2 * DH) * 2);
  _Float16* WrH   = (_Float16*)take((size_t)DH * DH * 2);
  _Float16* WcH   = (_Float16*)take((size_t)DH * DH * 2);
  _Float16* WpH   = (_Float16*)take((size_t)DIM * DH * 2);
  float* bqkv     = (float*)take((size_t)CQKV * 4);
  float* gscale   = (float*)take((size_t)(2 * DH) * 4);
  float* peRQ     = (float*)take((size_t)NH_KD * 64 * 4);
  float* peRK     = (float*)take((size_t)NH_KD * 64 * 4);
  float* peCQ     = (float*)take((size_t)NH_KD * 64 * 4);
  float* peCK     = (float*)take((size_t)NH_KD * 64 * 4);
  float* qkv      = (float*)take((size_t)BATCH * CQKV * HWSZ * 4);   // 134 MB
  float* rowm     = (float*)take((size_t)BATCH * CQKV * 64 * 4);
  float* colm     = (float*)take((size_t)BATCH * CQKV * 64 * 4);
  float* xr_pre   = (float*)take((size_t)BATCH * DH * 64 * 4);
  float* xc_pre   = (float*)take((size_t)BATCH * DH * 64 * 4);
  float* xr_fin   = (float*)take((size_t)BATCH * DH * 64 * 4);
  float* xc_fin   = (float*)take((size_t)BATCH * DH * 64 * 4);
  float* gate     = (float*)take((size_t)BATCH * DIM * HWSZ * 4);    // 67 MB

  // -------- 1) fold BN scales into weights, build aux tables --------
  auto blocks = [](int n) { return (n + 255) / 256; };
  k_foldW<<<blocks(NH_KD * DIM), 256, 0, stream>>>(WqkvH,                Wq, sq, NH_KD, DIM);
  k_foldW<<<blocks(NH_KD * DIM), 256, 0, stream>>>(WqkvH + NH_KD * DIM,  Wk, sk, NH_KD, DIM);
  k_foldW<<<blocks(DH * DIM),    256, 0, stream>>>(WqkvH + 2 * NH_KD * DIM, Wv, sv, DH, DIM);
  k_foldW<<<blocks(DIM * 2 * DH),256, 0, stream>>>(WpwH, Wpw, spw, DIM, 2 * DH);
  k_foldW<<<blocks(DH * DH),     256, 0, stream>>>(WrH, Wr, sr, DH, DH);
  k_foldW<<<blocks(DH * DH),     256, 0, stream>>>(WcH, Wc, sc, DH, DH);
  k_foldW<<<blocks(DIM * DH),    256, 0, stream>>>(WpH, Wp, sp, DIM, DH);
  k_copy<<<1, 256, 0, stream>>>(bqkv,              bq,  NH_KD);
  k_copy<<<1, 256, 0, stream>>>(bqkv + NH_KD,      bk_, NH_KD);
  k_copy<<<1, 256, 0, stream>>>(bqkv + 2 * NH_KD,  bv,  DH);
  k_mulvec<<<2, 256, 0, stream>>>(gscale, wdw, sdw, 2 * DH);
  k_pe<<<32, 256, 0, stream>>>(peRQ, pe_rq);
  k_pe<<<32, 256, 0, stream>>>(peRK, pe_rk);
  k_pe<<<32, 256, 0, stream>>>(peCQ, pe_cq);
  k_pe<<<32, 256, 0, stream>>>(peCK, pe_ck);

  // -------- 2) fused q/k/v GEMM: (B*4096 x 256) @ (256 x 512) --------
  k_gemm<<<dim3(BATCH * HWSZ / 64, CQKV / 128), 256, 0, stream>>>(
      x, (long long)DIM * HWSZ, DIM, HWSZ,
      nullptr, nullptr, 0, nullptr, nullptr,
      WqkvH, bqkv, CQKV, nullptr, qkv);

  // -------- 3) row/col means --------
  k_means<<<BATCH * CQKV, 256, 0, stream>>>(qkv, rowm, colm);

  // -------- 4) axial attention (row and column) --------
  k_axial<<<BATCH * HEADS, 128, 0, stream>>>(rowm, peRQ, peRK, xr_pre);
  k_axial<<<BATCH * HEADS, 128, 0, stream>>>(colm, peCQ, peCK, xc_pre);

  // -------- 5) proj_encode_row / proj_encode_column (ReLU input) --------
  k_gemm<<<dim3(BATCH, DH / 128), 256, 0, stream>>>(
      xr_pre, (long long)DH * 64, DH, 64,
      nullptr, nullptr, 1, nullptr, nullptr,
      WrH, br, DH, nullptr, xr_fin);
  k_gemm<<<dim3(BATCH, DH / 128), 256, 0, stream>>>(
      xc_pre, (long long)DH * 64, DH, 64,
      nullptr, nullptr, 1, nullptr, nullptr,
      WcH, bc, DH, nullptr, xc_fin);

  // -------- 6) gating branch: affine+ReLU on qkv, then 512->256 GEMM ------
  k_gemm<<<dim3(BATCH * HWSZ / 64, DIM / 128), 256, 0, stream>>>(
      qkv, (long long)CQKV * HWSZ, 2 * DH, HWSZ,
      gscale, bdw, 1, nullptr, nullptr,
      WpwH, bpw, DIM, nullptr, gate);

  // -------- 7) final: relu(v + xr + xc) @ Wp -> h_sigmoid -> * gate -------
  k_gemm<<<dim3(BATCH * HWSZ / 64, DIM / 128), 256, 0, stream>>>(
      qkv + (long long)(2 * NH_KD) * HWSZ,   // v region of batch 0
      (long long)CQKV * HWSZ, DH, HWSZ,
      nullptr, nullptr, 1, xr_fin, xc_fin,
      WpH, bp, DIM, gate, (float*)d_out);
}
